// RBFCluster_70901320122886
// MI455X (gfx1250) — 8-bit matrix operand order repaired and checked by reading, compiled, not run on hardware
//
#include <hip/hip_runtime.h>

typedef __attribute__((ext_vector_type(16))) int   i32x16_t;
typedef __attribute__((ext_vector_type(4)))  int   i32x4_t;
typedef __attribute__((ext_vector_type(2)))  int   i32x2_t;
typedef __attribute__((ext_vector_type(8)))  float f32x8_t;

typedef __attribute__((address_space(1))) i32x4_t* gbl_v4i_ptr;
typedef __attribute__((address_space(3))) i32x4_t* lds_v4i_ptr;

#define D_DIM   1024
#define C_DIM   2048
#define C_CHUNK 128
#define M_TILE  64
#define N_CHUNKS (C_DIM / C_CHUNK)

#define AB_SCALE      16.0f
#define W_EXP_MUL     (10.0f / 256.0f)
#define W_EXP_ADD     (-4.45482255564f)
#define OFF_SCALE     64.0f
#define OUT_UNSCALE   (1.0f / 16384.0f)

__device__ __forceinline__ f32x8_t wmma_fp8(i32x16_t a, i32x16_t b, f32x8_t c) {
  return __builtin_amdgcn_wmma_f32_16x16x128_fp8_fp8(a, b, (short)0, c, false, false);
}

__device__ __forceinline__ int pack4_fp8(float a, float b, float c, float d) {
  int p = __builtin_amdgcn_cvt_pk_fp8_f32(a, b, 0, false);
  p = __builtin_amdgcn_cvt_pk_fp8_f32(c, d, p, true);
  return p;
}

__device__ __forceinline__ unsigned char f32_to_fp8(float x) {
  return (unsigned char)(__builtin_amdgcn_cvt_pk_fp8_f32(x, 0.f, 0, false) & 0xff);
}

__device__ __forceinline__ i32x16_t load_a8(const unsigned char* row, int g) {
  i32x16_t A;
#pragma unroll
  for (int t = 0; t < 4; ++t) {
    i32x4_t v = *(const i32x4_t*)(row + 32 * t + 16 * g);
    A[4 * t] = v[0]; A[4 * t + 1] = v[1]; A[4 * t + 2] = v[2]; A[4 * t + 3] = v[3];
  }
  return A;
}

__device__ __forceinline__ i32x16_t load_b8(const unsigned char* col, int g) {
  i32x16_t B;
#pragma unroll
  for (int q = 0; q < 4; ++q) {
    i32x4_t v = *(const i32x4_t*)(col + 32 * q + 16 * g);
    B[4*q] = v[0]; B[4*q+1] = v[1]; B[4*q+2] = v[2]; B[4*q+3] = v[3];
  }
  return B;
}

__device__ __forceinline__ void async_copy_b128(unsigned lds_addr, const unsigned char* g) {
#if __has_builtin(__builtin_amdgcn_global_load_async_to_lds_b128)
  __builtin_amdgcn_global_load_async_to_lds_b128(
      (gbl_v4i_ptr)(uintptr_t)g,
      (lds_v4i_ptr)(uintptr_t)lds_addr, 0, 0);
#else
  unsigned long long ga = (unsigned long long)(uintptr_t)g;
  asm volatile("global_load_async_to_lds_b128 %0, %1, off" :: "v"(lds_addr), "v"(ga) : "memory");
#endif
}

__device__ __forceinline__ void wait_async0() {
#if __has_builtin(__builtin_amdgcn_s_wait_asynccnt)
  __builtin_amdgcn_s_wait_asynccnt(0);
#else
  asm volatile("s_wait_asynccnt 0x0" ::: "memory");
#endif
}

__device__ __forceinline__ unsigned lds_off32(const void* p) {
  return (unsigned)(uintptr_t)p;
}

__global__ __launch_bounds__(256) void prep_pos_kernel(const float* __restrict__ pos,
                                                       unsigned char* __restrict__ b_pre8) {
  __shared__ float red[256];
  const int c   = blockIdx.x;
  const int tid = threadIdx.x;
  const float* rp = pos + (size_t)c * D_DIM;
  float s = 0.f;
  for (int d = tid; d < D_DIM; d += 256) { float v = rp[d]; s += v * v; }
  red[tid] = s; __syncthreads();
  for (int o = 128; o > 0; o >>= 1) { if (tid < o) red[tid] += red[tid + o]; __syncthreads(); }
  const float inv = AB_SCALE / fmaxf(sqrtf(red[0]), 1e-12f);
  int* op = (int*)(b_pre8 + (size_t)c * D_DIM);
  const int d = tid * 4;
  op[tid] = pack4_fp8(rp[d] * inv, rp[d + 1] * inv, rp[d + 2] * inv, rp[d + 3] * inv);
}

__global__ __launch_bounds__(256) void prep_off_kernel(const float* __restrict__ offset,
                                                       unsigned char* __restrict__ offT8,
                                                       float* __restrict__ partials) {
  __shared__ float tile[32][33];
  __shared__ float red[256];
  const int tid = threadIdx.x;
  const int tx = tid & 31, ty = tid >> 5;
  const int c0 = blockIdx.x * 32, d0 = blockIdx.y * 32;
  float local = 0.f;
#pragma unroll
  for (int i = 0; i < 32; i += 8) {
    float v = offset[(size_t)(c0 + ty + i) * D_DIM + (d0 + tx)];
    tile[ty + i][tx] = v;
    local += v * v;
  }
  __syncthreads();
  {
    const int wx = (tid & 7) * 4;
    const int wy = tid >> 3;
    int* op = (int*)(offT8 + (size_t)(d0 + wy) * C_DIM + (c0 + wx));
    *op = pack4_fp8(tile[wx + 0][wy] * OFF_SCALE, tile[wx + 1][wy] * OFF_SCALE,
                    tile[wx + 2][wy] * OFF_SCALE, tile[wx + 3][wy] * OFF_SCALE);
  }
  red[tid] = local; __syncthreads();
  for (int o = 128; o > 0; o >>= 1) { if (tid < o) red[tid] += red[tid + o]; __syncthreads(); }
  if (tid == 0) partials[blockIdx.y * gridDim.x + blockIdx.x] = red[0];
}

__global__ __launch_bounds__(256) void reduce_partials_kernel(const float* __restrict__ partials,
                                                              int n, float* __restrict__ out_scalar) {
  __shared__ float red[256];
  float s = 0.f;
  for (int i = threadIdx.x; i < n; i += 256) s += partials[i];
  red[threadIdx.x] = s; __syncthreads();
  for (int o = 128; o > 0; o >>= 1) { if (threadIdx.x < o) red[threadIdx.x] += red[threadIdx.x + o]; __syncthreads(); }
  if (threadIdx.x == 0) *out_scalar = red[0];
}

__global__ __launch_bounds__(512) void rbf_main_kernel(const float* __restrict__ inp,
                                                       const unsigned char* __restrict__ b_pre8,
                                                       const unsigned char* __restrict__ offT8,
                                                       float* __restrict__ out) {
  __shared__ __align__(16) unsigned char aTile[M_TILE][D_DIM];
  __shared__ __align__(16) unsigned char bTile[C_CHUNK][D_DIM];
  __shared__ __align__(16) unsigned char wTile[M_TILE][C_CHUNK];
  __shared__ float rowPart[M_TILE][8];
  __shared__ float rowInv[M_TILE];

  const int tid  = threadIdx.x;
  const int lane = tid & 31;
  const int wave = tid >> 5;
  const int g    = lane >> 4;
  const int m    = lane & 15;
  const int r    = wave >> 2;
  const int h    = wave & 3;
  const int rowBase = blockIdx.x * M_TILE;

  {
    const unsigned char* src = b_pre8;
    const unsigned bbase = lds_off32(&bTile[0][0]);
#pragma unroll
    for (int it = 0; it < 16; ++it) {
      const int off = tid * 16 + it * 8192;
      async_copy_b128(bbase + off, src + off);
    }
  }

  {
    const int rr = tid >> 3;
    const int t  = tid & 7;
    const float* rp = inp + (size_t)(rowBase + rr) * D_DIM;
    float s = 0.f;
    for (int d = t; d < D_DIM; d += 8) { float v = rp[d]; s += v * v; }
    rowPart[rr][t] = s;
  }
  __syncthreads();
  if (tid < M_TILE) {
    float s = 0.f;
#pragma unroll
    for (int i = 0; i < 8; ++i) s += rowPart[tid][i];
    rowInv[tid] = AB_SCALE / fmaxf(sqrtf(s), 1e-12f);
  }
  __syncthreads();
  {
    int* aInt = (int*)aTile;
    for (int i = tid; i < M_TILE * D_DIM / 4; i += 512) {
      const int rr = i >> 8;
      const int d  = (i * 4) & (D_DIM - 1);
      const float* rp = inp + (size_t)(rowBase + rr) * D_DIM + d;
      const float inv = rowInv[rr];
      aInt[i] = pack4_fp8(rp[0] * inv, rp[1] * inv, rp[2] * inv, rp[3] * inv);
    }
  }

  f32x8_t acc[16];
#pragma unroll
  for (int t = 0; t < 16; ++t)
#pragma unroll
    for (int e = 0; e < 8; ++e) acc[t][e] = 0.f;

  const unsigned char* aRow = &aTile[r * 16 + m][0];
  const unsigned char* offBase0 = offT8 + (size_t)(h * 256 + m) * C_DIM;

  for (int ch = 0; ch < N_CHUNKS; ++ch) {
    const int c0 = ch * C_CHUNK;

    wait_async0();
    __syncthreads();

    f32x8_t s0, s1;
#pragma unroll
    for (int e = 0; e < 8; ++e) { s0[e] = 0.f; s1[e] = 0.f; }
    const unsigned char* bRow0 = &bTile[h * 32 + m][0];
    const unsigned char* bRow1 = &bTile[h * 32 + 16 + m][0];
#pragma unroll
    for (int k0 = 0; k0 < D_DIM; k0 += 128) {
      i32x16_t A  = load_a8(aRow + k0, g);
      i32x16_t B0 = load_b8(bRow0 + k0, g);
      s0 = wmma_fp8(A, B0, s0);
      i32x16_t B1 = load_b8(bRow1 + k0, g);
      s1 = wmma_fp8(A, B1, s1);
    }
#pragma unroll
    for (int j = 0; j < 8; ++j) {
      const float w0 = __expf(fmaf(s0[j], W_EXP_MUL, W_EXP_ADD));
      wTile[r * 16 + j + 8 * g][h * 32 + m] = f32_to_fp8(w0);
      const float w1 = __expf(fmaf(s1[j], W_EXP_MUL, W_EXP_ADD));
      wTile[r * 16 + j + 8 * g][h * 32 + 16 + m] = f32_to_fp8(w1);
    }
    __syncthreads();

    if (ch + 1 < N_CHUNKS) {
      const unsigned char* src = b_pre8 + (size_t)(c0 + C_CHUNK) * D_DIM;
      const unsigned bbase = lds_off32(&bTile[0][0]);
#pragma unroll
      for (int it = 0; it < 16; ++it) {
        const int off = tid * 16 + it * 8192;
        async_copy_b128(bbase + off, src + off);
      }
    }

    {
      i32x16_t A = load_a8(&wTile[r * 16 + m][0], g);
      const unsigned char* offBase = offBase0 + c0;
      i32x16_t Bcur = load_b8(offBase, g);
#pragma unroll
      for (int t = 0; t < 16; ++t) {
        i32x16_t Bnxt;
        if (t < 15) Bnxt = load_b8(offBase + (size_t)(t + 1) * 16 * C_DIM, g);
        acc[t] = wmma_fp8(A, Bcur, acc[t]);
        Bcur = Bnxt;
      }
    }
    __syncthreads();
  }

  {
    float* outBase = out + (size_t)(rowBase + r * 16 + 8 * g) * D_DIM + h * 256 + m;
    const float* inBase = inp + (size_t)(rowBase + r * 16 + 8 * g) * D_DIM + h * 256 + m;
#pragma unroll
    for (int t = 0; t < 16; ++t)
#pragma unroll
      for (int j = 0; j < 8; ++j)
        outBase[(size_t)j * D_DIM + t * 16] =
            fmaf(acc[t][j], OUT_UNSCALE, inBase[(size_t)j * D_DIM + t * 16]);
  }
}

__global__ __launch_bounds__(256) void normalize_rows_kernel(float* __restrict__ out) {
  __shared__ float red[256];
  const int tid = threadIdx.x;
  float* rp = out + (size_t)blockIdx.x * D_DIM;
  float s = 0.f;
  for (int d = tid; d < D_DIM; d += 256) { float v = rp[d]; s += v * v; }
  red[tid] = s; __syncthreads();
  for (int o = 128; o > 0; o >>= 1) { if (tid < o) red[tid] += red[tid + o]; __syncthreads(); }
  const float inv = 1.f / fmaxf(sqrtf(red[0]), 1e-12f);
  for (int d = tid; d < D_DIM; d += 256) rp[d] *= inv;
}

extern "C" void kernel_launch(void* const* d_in, const int* in_sizes, int n_in,
                              void* d_out, int out_size, void* d_ws, size_t ws_size,
                              hipStream_t stream) {
  const float* inp    = (const float*)d_in[0];
  const float* pos    = (const float*)d_in[1];
  const float* offset = (const float*)d_in[2];
  float* out = (float*)d_out;

  const int N = in_sizes[0] / D_DIM;

  unsigned char* b_pre8 = (unsigned char*)d_ws;
  unsigned char* offT8  = (unsigned char*)d_ws + (size_t)C_DIM * D_DIM;
  float*         partial = (float*)((unsigned char*)d_ws + 2 * (size_t)C_DIM * D_DIM);

  prep_pos_kernel<<<C_DIM, 256, 0, stream>>>(pos, b_pre8);

  dim3 tg(C_DIM / 32, D_DIM / 32);
  prep_off_kernel<<<tg, 256, 0, stream>>>(offset, offT8, partial);
  reduce_partials_kernel<<<1, 256, 0, stream>>>(partial, (C_DIM / 32) * (D_DIM / 32),
                                                out + (size_t)N * D_DIM);

  rbf_main_kernel<<<N / M_TILE, 512, 0, stream>>>(inp, b_pre8, offT8, out);
  normalize_rows_kernel<<<N, 256, 0, stream>>>(out);
}
